// LSTMHead_29291676959159
// MI455X (gfx1250) — compile-verified, matrix operand order checked by reading, not run on hardware
//
#include <hip/hip_runtime.h>
#include <hip/hip_bf16.h>

// ---------------------------------------------------------------------------
// Types for CDNA5 WMMA (wave32): v_wmma_f32_16x16x32_bf16
// ---------------------------------------------------------------------------
typedef __attribute__((ext_vector_type(8)))  __bf16 v8bf;
typedef __attribute__((ext_vector_type(16))) __bf16 v16bf;
typedef __attribute__((ext_vector_type(8)))  float  v8f;

#define BB 32      // batch
#define TT 1024    // seq len
#define FF 2048    // feature
#define HH 512     // hidden
#define GG 2048    // 4*H
#define OO 7       // out size
#define NWG_SCAN 16
#define HSLICE 32  // H columns per scan workgroup

__device__ __forceinline__ v8f wmma_bf16(v16bf a, v16bf b, v8f c) {
  // 8 args: (neg_a, A, neg_b, B, c_mod, C, reuse_a, reuse_b)
  return __builtin_amdgcn_wmma_f32_16x16x32_bf16(false, a, false, b, (short)0, c,
                                                 false, false);
}

// CDNA5 async global->LDS copy (ASYNCcnt path, CDNA5 ISA).
// lds_off: per-lane LDS byte address (low 32 bits of generic LDS pointer);
// gptr:    per-lane 64-bit global address.
__device__ __forceinline__ void async_load_b128(unsigned lds_off,
                                                const void* gptr) {
  asm volatile("global_load_async_to_lds_b128 %0, %1, off"
               :
               : "v"(lds_off), "v"(gptr)
               : "memory");
}
__device__ __forceinline__ void wait_asynccnt0() {
  asm volatile("s_wait_asynccnt 0x0" ::: "memory");
}
__device__ __forceinline__ unsigned lds_off_of(const void* p) {
  return (unsigned)(size_t)p;  // LDS address is the pointer's low 32 bits
}

// Load a 16x32 bf16 fragment (A layout; B mirrors with n=lane%16) from a
// row-major [rows][ld] bf16 array.  Per ISA 7.12.2 (wave32):
//   lane<16 : element m = lane,    K = {k0..k0+7} U {k0+16..k0+23}
//   lane>=16: element m = lane-16, K = {k0+8..k0+15} U {k0+24..k0+31}
// i.e. two contiguous 16-byte chunks per lane -> 2x ds_load_b128.
__device__ __forceinline__ v16bf load_frag(const __bf16* base, int row0, int k0,
                                           int ld, int lane) {
  int m  = lane & 15;
  int kb = k0 + ((lane & 16) ? 8 : 0);
  const __bf16* p = base + (size_t)(row0 + m) * ld + kb;
  v8bf lo = *(const v8bf*)(p);
  v8bf hi = *(const v8bf*)(p + 16);
  v16bf r;
#pragma unroll
  for (int i = 0; i < 8; ++i) { r[i] = lo[i]; r[i + 8] = hi[i]; }
  return r;
}

__device__ __forceinline__ float sigmoidf_fast(float x) {
  return 1.0f / (1.0f + __expf(-x));
}

// ---------------------------------------------------------------------------
// fp32 -> bf16 conversion (grid-stride)
// ---------------------------------------------------------------------------
__global__ __launch_bounds__(256) void f32_to_bf16_kernel(
    const float* __restrict__ in, __bf16* __restrict__ out, size_t n) {
  size_t i = (size_t)blockIdx.x * blockDim.x + threadIdx.x;
  size_t stride = (size_t)gridDim.x * blockDim.x;
  for (; i < n; i += stride) out[i] = (__bf16)in[i];
}

// ---------------------------------------------------------------------------
// Phase 1: xg[32768,2048] = x_bf16[32768,2048] @ W_ih_bf16[2048,2048]^T + bias
// 128x128 tile per WG, 8 waves (2Mx4N tiles each).  Double-buffered LDS tiles
// filled by async global->LDS copies: loads for k+1 overlap WMMAs for k.
// ---------------------------------------------------------------------------
__global__ __launch_bounds__(256) void gemm_xg_kernel(
    const __bf16* __restrict__ X,   // [32768, 2048]
    const __bf16* __restrict__ W,   // [2048(g), 2048(f)]
    const float* __restrict__ b_ih, const float* __restrict__ b_hh,
    float* __restrict__ XG) {       // [32768, 2048]
  const int K = FF;
  const int LDT = 40;               // 32 + 8 pad halfs (keeps 16B align)
  __shared__ __bf16 As[2][128 * LDT];
  __shared__ __bf16 Bs[2][128 * LDT];

  const int m0 = blockIdx.y * 128;
  const int n0 = blockIdx.x * 128;
  const int tid = threadIdx.x;
  const int lane = tid & 31;
  const int wave = tid >> 5;       // 0..7
  const int mg = wave & 3;         // 4 groups of 2 M-tiles
  const int ng = wave >> 2;        // 2 groups of 4 N-tiles

  v8f acc[2][4];
#pragma unroll
  for (int mi = 0; mi < 2; ++mi)
#pragma unroll
    for (int ni = 0; ni < 4; ++ni) acc[mi][ni] = (v8f){};

  // issue async copies of one 128x32 A tile + B tile into buffer `buf`
  auto issue_tile = [&](int k0, int buf) {
#pragma unroll
    for (int c = tid; c < 512; c += 256) {
      int row = c >> 2;
      int cp  = (c & 3) * 8;
      async_load_b128(lds_off_of(&As[buf][row * LDT + cp]),
                      &X[(size_t)(m0 + row) * K + k0 + cp]);
      async_load_b128(lds_off_of(&Bs[buf][row * LDT + cp]),
                      &W[(size_t)(n0 + row) * K + k0 + cp]);
    }
  };

  issue_tile(0, 0);
  wait_asynccnt0();
  __syncthreads();

  for (int it = 0; it < K / 32; ++it) {
    const int buf = it & 1;
    if (it + 1 < K / 32) issue_tile((it + 1) * 32, buf ^ 1);

    v16bf a0 = load_frag(As[buf], (mg * 2 + 0) * 16, 0, LDT, lane);
    v16bf a1 = load_frag(As[buf], (mg * 2 + 1) * 16, 0, LDT, lane);
#pragma unroll
    for (int ni = 0; ni < 4; ++ni) {
      v16bf bf = load_frag(Bs[buf], (ng * 4 + ni) * 16, 0, LDT, lane);
      acc[0][ni] = wmma_bf16(a0, bf, acc[0][ni]);
      acc[1][ni] = wmma_bf16(a1, bf, acc[1][ni]);
    }
    wait_asynccnt0();
    __syncthreads();
  }

  // Epilogue: C layout (wave32): VGPR r -> M = r + (lane>=16 ? 8 : 0), N = lane&15
#pragma unroll
  for (int mi = 0; mi < 2; ++mi) {
#pragma unroll
    for (int ni = 0; ni < 4; ++ni) {
      int mrow = m0 + (mg * 2 + mi) * 16 + ((lane & 16) ? 8 : 0);
      int ncol = n0 + (ng * 4 + ni) * 16 + (lane & 15);
      float bias = b_ih[ncol] + b_hh[ncol];
#pragma unroll
      for (int r = 0; r < 8; ++r)
        XG[(size_t)(mrow + r) * GG + ncol] = acc[mi][ni][r] + bias;
    }
  }
}

// ---------------------------------------------------------------------------
// Scan init: zero h double-buffer[0] and the grid-barrier counter
// ---------------------------------------------------------------------------
__global__ __launch_bounds__(256) void scan_init_kernel(__bf16* hbuf,
                                                        unsigned* counter) {
  int i = blockIdx.x * blockDim.x + threadIdx.x;
  if (i < BB * HH) hbuf[i] = (__bf16)0.0f;
  if (i == 0) *counter = 0u;
}

// ---------------------------------------------------------------------------
// Phase 2: persistent LSTM scan.  16 WGs, WG w owns h columns [32w, 32w+32).
// Its 128 W_hh rows (4 gates x 32 cols) stay in LDS (bf16, 128KB) for all
// 1024 steps (only possible with CDNA5's 320KB/WGP).  Per step: async-load
// full h^T (32KB bf16) -> LDS, 256 WMMAs for the [128x32] gate slice,
// activations + c/h update, write h slice to the other global h buffer, one
// grid barrier (double-buffered h makes a single barrier per step safe).
// ---------------------------------------------------------------------------
__global__ __launch_bounds__(256) void lstm_scan_kernel(
    const float* __restrict__ XG,     // [B, T, 4H]
    const __bf16* __restrict__ Whh,   // [4H, H] bf16 row-major
    __bf16* __restrict__ hbuf,        // [2, B, H] bf16 double buffer
    float* __restrict__ HS,           // [B, T, H]
    unsigned* __restrict__ counter) {
  const int LDW = 520;                // 512 + 8 pad halfs
  extern __shared__ __align__(16) char smem[];
  __bf16* Wl = (__bf16*)smem;                       // [128][520]  133120 B
  __bf16* Hl = Wl + 128 * LDW;                      // [32][520]    33280 B
  float*  Gl = (float*)(Hl + 32 * LDW);             // [128][33]    16896 B
  float*  Cl = Gl + 128 * 33;                       // [32][33]      4224 B

  const int wg = blockIdx.x;
  const int j0 = wg * HSLICE;
  const int tid = threadIdx.x;
  const int lane = tid & 31;
  const int wave = tid >> 5;        // 0..7 == M-tile index

  // Preload this WG's W_hh slice via async copies: LDS row lr = gate*32 + r
  // <- global row gate*512 + j0 + r.  64 x 16B chunks per row.
  for (int c = tid; c < 128 * 64; c += 256) {
    int lr = c >> 6;
    int cp = (c & 63) * 8;
    int g = lr >> 5, r = lr & 31;
    async_load_b128(lds_off_of(&Wl[lr * LDW + cp]),
                    &Whh[(size_t)(g * HH + j0 + r) * HH + cp]);
  }
  for (int i = tid; i < 32 * 33; i += 256) Cl[i] = 0.0f;
  wait_asynccnt0();
  __syncthreads();

  for (int t = 0; t < TT; ++t) {
    // ---- async-load h^T (batch-major, K contiguous) for this step ----
    const __bf16* hin = hbuf + (size_t)(t & 1) * BB * HH;
    for (int c = tid; c < 32 * 64; c += 256) {
      int b = c >> 6;
      int cp = (c & 63) * 8;
      async_load_b128(lds_off_of(&Hl[b * LDW + cp]), &hin[b * HH + cp]);
    }
    wait_asynccnt0();
    __syncthreads();

    // ---- WMMA: gate_slice[128,32] = W_slice[128,512] x h^T[512,32] ----
    v8f acc0 = (v8f){}, acc1 = (v8f){};
#pragma unroll
    for (int kk = 0; kk < 16; ++kk) {
      v16bf a  = load_frag(Wl, wave * 16, kk * 32, LDW, lane);
      v16bf b0 = load_frag(Hl, 0,         kk * 32, LDW, lane);
      v16bf b1 = load_frag(Hl, 16,        kk * 32, LDW, lane);
      acc0 = wmma_bf16(a, b0, acc0);
      acc1 = wmma_bf16(a, b1, acc1);
    }
    // scatter C tiles into gates LDS: row = wave*16 + r + (lane>=16 ? 8 : 0)
    {
      int lr = wave * 16 + ((lane & 16) ? 8 : 0);
      int bc = lane & 15;
#pragma unroll
      for (int r = 0; r < 8; ++r) {
        Gl[(lr + r) * 33 + bc]      = acc0[r];
        Gl[(lr + r) * 33 + 16 + bc] = acc1[r];
      }
    }
    __syncthreads();

    // ---- activations + state update for 32 cols x 32 batch ----
    for (int i = tid; i < 32 * 32; i += 256) {
      int b = i >> 5;
      int j = i & 31;
      const float* xgp = &XG[((size_t)b * TT + t) * GG + j0 + j];
      // speculative prefetch of next timestep's xg slice (global_prefetch_b8)
      __builtin_prefetch(xgp + GG, 0, 1);
      float gi = Gl[(0 * 32 + j) * 33 + b] + xgp[0];
      float gf = Gl[(1 * 32 + j) * 33 + b] + xgp[512];
      float gg = Gl[(2 * 32 + j) * 33 + b] + xgp[1024];
      float go = Gl[(3 * 32 + j) * 33 + b] + xgp[1536];
      float ig = sigmoidf_fast(gi);
      float fg = sigmoidf_fast(gf);
      float gc = tanhf(gg);
      float og = sigmoidf_fast(go);
      float cn = fg * Cl[j * 33 + b] + ig * gc;
      float hn = og * tanhf(cn);
      Cl[j * 33 + b] = cn;
      HS[((size_t)b * TT + t) * HH + j0 + j] = hn;
      hbuf[(size_t)((t + 1) & 1) * BB * HH + (size_t)b * HH + j0 + j] =
          (__bf16)hn;
    }

    // ---- one grid barrier per step ----
    __threadfence();
    __syncthreads();
    if (tid == 0) {
      atomicAdd(counter, 1u);
      unsigned target = (unsigned)gridDim.x * (unsigned)(t + 1);
      while (__hip_atomic_load(counter, __ATOMIC_RELAXED,
                               __HIP_MEMORY_SCOPE_AGENT) < target)
        __builtin_amdgcn_s_sleep(8);
    }
    __syncthreads();
  }
}

// ---------------------------------------------------------------------------
// Phase 3: out[bt,o] = HS[bt,:] . W_out[o,:] + b_out[o]; one wave per row.
// ---------------------------------------------------------------------------
__global__ __launch_bounds__(256) void out_proj_kernel(
    const float* __restrict__ HS, const float* __restrict__ Wout,
    const float* __restrict__ bout, float* __restrict__ OUT) {
  int bt = blockIdx.x * 8 + (threadIdx.x >> 5);
  int lane = threadIdx.x & 31;
  const float* hrow = &HS[(size_t)bt * HH];
  float hv[16];
#pragma unroll
  for (int i = 0; i < 16; ++i) hv[i] = hrow[lane + 32 * i];
#pragma unroll
  for (int o = 0; o < OO; ++o) {
    float s = 0.0f;
#pragma unroll
    for (int i = 0; i < 16; ++i) s += hv[i] * Wout[o * HH + lane + 32 * i];
#pragma unroll
    for (int off = 16; off >= 1; off >>= 1) s += __shfl_xor(s, off, 32);
    if (lane == 0) OUT[(size_t)bt * OO + o] = s + bout[o];
  }
}

// ---------------------------------------------------------------------------
extern "C" void kernel_launch(void* const* d_in, const int* in_sizes, int n_in,
                              void* d_out, int out_size, void* d_ws,
                              size_t ws_size, hipStream_t stream) {
  (void)in_sizes; (void)n_in; (void)out_size; (void)ws_size;
  const float* x     = (const float*)d_in[0];
  const float* W_ih  = (const float*)d_in[1];
  const float* W_hh  = (const float*)d_in[2];
  const float* b_ih  = (const float*)d_in[3];
  const float* b_hh  = (const float*)d_in[4];
  const float* W_out = (const float*)d_in[5];
  const float* b_out = (const float*)d_in[6];
  float* out = (float*)d_out;

  // workspace carve-up (256B aligned)
  char* ws = (char*)d_ws;
  size_t off = 0;
  auto carve = [&](size_t bytes) {
    void* p = ws + off;
    off = (off + bytes + 255) & ~(size_t)255;
    return p;
  };
  float*   xg      = (float*)carve((size_t)BB * TT * GG * 4);   // 268 MB
  __bf16*  x_bf    = (__bf16*)carve((size_t)BB * TT * FF * 2);  // 134 MB
  __bf16*  wih_bf  = (__bf16*)carve((size_t)GG * FF * 2);       //   8 MB
  __bf16*  whh_bf  = (__bf16*)carve((size_t)GG * HH * 2);       //   2 MB
  float*   hs      = (float*)carve((size_t)BB * TT * HH * 4);   //  67 MB
  __bf16*  hbuf    = (__bf16*)carve((size_t)2 * BB * HH * 2);   //  64 KB
  unsigned* counter = (unsigned*)carve(256);

  const int scan_smem = (128 * 520 + 32 * 520) * 2 + (128 * 33 + 32 * 33) * 4;
  hipFuncSetAttribute((const void*)lstm_scan_kernel,
                      hipFuncAttributeMaxDynamicSharedMemorySize, scan_smem);

  // 1) bf16 conversions
  f32_to_bf16_kernel<<<4096, 256, 0, stream>>>(x, x_bf, (size_t)BB * TT * FF);
  f32_to_bf16_kernel<<<1024, 256, 0, stream>>>(W_ih, wih_bf, (size_t)GG * FF);
  f32_to_bf16_kernel<<<256, 256, 0, stream>>>(W_hh, whh_bf, (size_t)GG * HH);

  // 2) xg = x @ W_ih^T + (b_ih + b_hh)   [WMMA bf16, f32 accum, async tiles]
  gemm_xg_kernel<<<dim3(GG / 128, (BB * TT) / 128), 256, 0, stream>>>(
      x_bf, wih_bf, b_ih, b_hh, xg);

  // 3) recurrent scan (persistent, grid barrier per step)
  scan_init_kernel<<<(BB * HH + 255) / 256, 256, 0, stream>>>(hbuf, counter);
  lstm_scan_kernel<<<NWG_SCAN, 256, scan_smem, stream>>>(xg, whh_bf, hbuf, hs,
                                                         counter);

  // 4) output projection
  out_proj_kernel<<<(BB * TT) / 8, 256, 0, stream>>>(hs, W_out, b_out, out);
}
